// GraphConvolution_53180285059793
// MI455X (gfx1250) — matrix operand order repaired and checked by reading, not compiled, not run on hardware
//
#include <hip/hip_runtime.h>
#include <hip/hip_bf16.h>

// CDNA5 (gfx1250) wave32 WMMA GEMM, bf16 inputs / f32 accumulate.
// 128x256 block tile, 64x64 wave tile, double-buffered LDS pipeline,
// transposed bf16 intermediate so the big GEMM needs no LDS scatter,
// L2-cooperative blocking (4 column-blocks share each adj row strip).
typedef __attribute__((ext_vector_type(16))) __bf16 v16bf;
typedef __attribute__((ext_vector_type(8)))  float  v8f;
typedef __attribute__((ext_vector_type(4)))  float  v4f;

// pack two f32 -> two bf16 in one dword (x -> low, y -> high)
#if __has_builtin(__builtin_amdgcn_cvt_pk_bf16_f32)
__device__ __forceinline__ unsigned pk2bf(float x, float y) {
  auto p = __builtin_amdgcn_cvt_pk_bf16_f32(x, y);
  return __builtin_bit_cast(unsigned, p);
}
#elif __has_builtin(__builtin_amdgcn_perm)
__device__ __forceinline__ unsigned pk2bf(float x, float y) {
  // round-to-nearest (ties up) then truncate via byte-permute: 3 VALU / 2 elems
  unsigned a = __builtin_bit_cast(unsigned, x) + 0x8000u;
  unsigned b = __builtin_bit_cast(unsigned, y) + 0x8000u;
  return __builtin_amdgcn_perm(b, a, 0x07060302u);  // bytes: y3 y2 x3 x2
}
#else
__device__ __forceinline__ unsigned pk2bf(float x, float y) {
  unsigned a = __builtin_bit_cast(unsigned, x) + 0x8000u;
  unsigned b = __builtin_bit_cast(unsigned, y) + 0x8000u;
  return (a >> 16) | (b & 0xFFFF0000u);
}
#endif

struct Frag32 { int4 lo, hi; };  // 32 bytes == v16bf

// C = A[M,K](f32) * B [+ bias]
//  - B_T_BF16=0: B is f32 [K,N] row-major (transposed into LDS with scatter)
//  - B_T_BF16=1: B is bf16 [N,K] row-major (stride K) -> contiguous LDS copy
//  - OUT_T_BF16=1: C written bf16 TRANSPOSED as [N,M] (stride M), packed b128 stores
template <bool B_T_BF16, bool OUT_T_BF16, bool HAS_BIAS>
__global__ __launch_bounds__(256)
void gemm_bf16_wmma(const float* __restrict__ A,
                    const void*  __restrict__ Bv,
                    const float* __restrict__ bias,
                    void*        __restrict__ Cv,
                    int M, int N, int K) {
  constexpr int BM = 128, BN = 256, BK = 32;
  constexpr int AS = BK + 8;  // padded LDS strides (bf16 units)
  constexpr int BS = BK + 8;

  __shared__ unsigned short sA[2][BM * AS];  // A tile [m][k] bf16 (20 KB)
  __shared__ unsigned short sB[2][BN * BS];  // B tile [n][k] bf16 (40 KB)

  const int tid  = threadIdx.x;
  const int lane = tid & 31;
  const int wave = tid >> 5;          // 0..7
  const int wm   = wave & 1;          // 2 row-groups of 64
  const int wn   = wave >> 1;         // 4 col-groups of 64
  const int lrow = lane & 15;
  const int half = lane >> 4;         // 0 / 1

  const int rowBase = blockIdx.y * BM;
  const int colBase = blockIdx.x * BN;

  const float*          Bf = (const float*)Bv;
  const unsigned short* Bh = (const unsigned short*)Bv;

  v8f acc[4][4];                      // 64x64 wave tile = 16 WMMA accumulators
#pragma unroll
  for (int mi = 0; mi < 4; ++mi)
#pragma unroll
    for (int ni = 0; ni < 4; ++ni) {
      v8f z{};
      acc[mi][ni] = z;
    }

  // ---- staging registers (next tile in flight) ----
  v4f   aR[4];
  v4f   bRf[8];
  uint4 bRh[4];

  auto loadA = [&](int k0) {
#pragma unroll
    for (int i = 0; i < 4; ++i) {
      int f  = tid + i * 256;      // 1024 float4 slots (128 x 32 f32)
      int r  = f >> 3;
      int kc = (f & 7) << 2;
      aR[i] = *(const v4f*)(A + (long)(rowBase + r) * K + k0 + kc);
    }
  };
  auto loadB = [&](int k0) {
    if constexpr (B_T_BF16) {
      // B is [N,K] bf16: rows of the LDS tile are contiguous in memory
#pragma unroll
      for (int i = 0; i < 4; ++i) {
        int f  = tid + i * 256;    // 1024 slots of 8 bf16 (256 x 32 bf16)
        int n  = f >> 2;           // 0..255
        int kc = (f & 3) << 3;     // 0,8,16,24
        bRh[i] = *(const uint4*)(Bh + (long)(colBase + n) * K + k0 + kc);
      }
    } else {
#pragma unroll
      for (int i = 0; i < 8; ++i) {
        int f  = tid + i * 256;    // 2048 float4 slots (32 x 256 f32)
        int kr = f >> 6;
        int nc = (f & 63) << 2;
        bRf[i] = *(const v4f*)(Bf + (long)(k0 + kr) * N + colBase + nc);
      }
    }
  };
  auto stage = [&](int buf) {
    unsigned short* __restrict__ pA = sA[buf];
    unsigned short* __restrict__ pB = sB[buf];
#pragma unroll
    for (int i = 0; i < 4; ++i) {
      int f  = tid + i * 256;
      int r  = f >> 3;
      int kc = (f & 7) << 2;
      unsigned* d = (unsigned*)(pA + r * AS + kc);
      d[0] = pk2bf(aR[i].x, aR[i].y);
      d[1] = pk2bf(aR[i].z, aR[i].w);
    }
    if constexpr (B_T_BF16) {
#pragma unroll
      for (int i = 0; i < 4; ++i) {
        int f  = tid + i * 256;
        int n  = f >> 2;
        int kc = (f & 3) << 3;
        *(uint4*)(pB + n * BS + kc) = bRh[i];   // straight b128 copy
      }
    } else {
      // f32 B: convert + transpose-scatter (only used by the small GEMM1)
#pragma unroll
      for (int i = 0; i < 8; ++i) {
        int f  = tid + i * 256;
        int kr = f >> 6;
        int nc = (f & 63) << 2;
        unsigned u01 = pk2bf(bRf[i].x, bRf[i].y);
        unsigned u23 = pk2bf(bRf[i].z, bRf[i].w);
        pB[(nc + 0) * BS + kr] = (unsigned short)u01;
        pB[(nc + 1) * BS + kr] = (unsigned short)(u01 >> 16);
        pB[(nc + 2) * BS + kr] = (unsigned short)u23;
        pB[(nc + 3) * BS + kr] = (unsigned short)(u23 >> 16);
      }
    }
  };
  auto compute = [&](int buf) {
    const unsigned short* __restrict__ pA = sA[buf];
    const unsigned short* __restrict__ pB = sB[buf];
    // A frags (ISA 16-bit A 16x32 layout): lanes0-15 K0-7/16-23, lanes16-31 K8-15/24-31
    v16bf af[4];
#pragma unroll
    for (int mi = 0; mi < 4; ++mi) {
      int r  = wm * 64 + mi * 16 + lrow;
      int ko = half * 8;
      Frag32 fr;
      fr.lo = *(const int4*)(pA + r * AS + ko);
      fr.hi = *(const int4*)(pA + r * AS + ko + 16);
      af[mi] = __builtin_bit_cast(v16bf, fr);
    }
    // B frags: lanes0-15 col=lane K0-7/16-23, lanes16-31 K8-15/24-31
#pragma unroll
    for (int ni = 0; ni < 4; ++ni) {
      int c  = wn * 64 + ni * 16 + lrow;
      int ko = half * 8;
      Frag32 fr;
      fr.lo = *(const int4*)(pB + c * BS + ko);
      fr.hi = *(const int4*)(pB + c * BS + ko + 16);
      v16bf bfr = __builtin_bit_cast(v16bf, fr);
#pragma unroll
      for (int mi = 0; mi < 4; ++mi) {
        acc[mi][ni] = __builtin_amdgcn_wmma_f32_16x16x32_bf16(
            false, af[mi], false, bfr, (short)0, acc[mi][ni], false, false);
      }
    }
  };

  // ---- software pipeline: load(k+1) || compute(k), one barrier per step ----
  loadA(0);
  loadB(0);
  stage(0);
  __syncthreads();

  int cur = 0;
  for (int k0 = 0; k0 < K; k0 += BK) {
    const int kn = k0 + BK;
    if (kn < K) {           // issue next tile's global loads early
      loadA(kn);
      loadB(kn);
    }
    compute(cur);           // ds_load + wmma hide the in-flight global loads
    if (kn < K)
      stage(cur ^ 1);       // wait loads, convert, ds_store into other buffer
    __syncthreads();
    cur ^= 1;
  }

  // ---- epilogue: lane half h -> VGPR r holds C[m16 + h*8 + r][n16 + lrow]
#pragma unroll
  for (int mi = 0; mi < 4; ++mi) {
#pragma unroll
    for (int ni = 0; ni < 4; ++ni) {
      int row0 = rowBase + wm * 64 + mi * 16 + half * 8;
      int col  = colBase + wn * 64 + ni * 16 + lrow;
      if constexpr (OUT_T_BF16) {
        // transposed bf16 output [N,M]: the 8 rows are contiguous -> one b128
        v8f a = acc[mi][ni];
        uint4 o;
        o.x = pk2bf(a[0], a[1]);
        o.y = pk2bf(a[2], a[3]);
        o.z = pk2bf(a[4], a[5]);
        o.w = pk2bf(a[6], a[7]);
        *(uint4*)((unsigned short*)Cv + (long)col * M + row0) = o;
      } else {
        float badd = HAS_BIAS ? bias[col] : 0.0f;
#pragma unroll
        for (int r = 0; r < 8; ++r) {
          float v = acc[mi][ni][r] + badd;
          ((float*)Cv)[(long)(row0 + r) * N + col] = v;
        }
      }
    }
  }
}

extern "C" void kernel_launch(void* const* d_in, const int* in_sizes, int n_in,
                              void* d_out, int out_size, void* d_ws, size_t ws_size,
                              hipStream_t stream) {
  // setup_inputs order: x [8192,1024] f32, adj [8192,8192] f32,
  //                     weight [1024,1024] f32, bias [1024] f32
  const float* x    = (const float*)d_in[0];
  const float* adj  = (const float*)d_in[1];
  const float* wgt  = (const float*)d_in[2];
  const float* bias = (const float*)d_in[3];

  constexpr int Nrows = 8192, IN_F = 1024, OUT_F = 1024;

  // supportT = (x @ weight)^T, bf16 [OUT_F][Nrows] in workspace (16 MB, L2-hot)
  unsigned short* supportT = (unsigned short*)d_ws;

  dim3 block(256);
  dim3 grid(OUT_F / 256, Nrows / 128);  // (4, 64); x-major => adj strip sharers adjacent

  // GEMM1: supportT(bf16,[N,M]) = x @ weight     (B = f32 [K,N])
  gemm_bf16_wmma<false, true, false><<<grid, block, 0, stream>>>(
      x, (const void*)wgt, nullptr, (void*)supportT, Nrows, OUT_F, IN_F);

  // GEMM2: out(f32) = adj @ support + bias   (B = bf16 [N,K] = supportT, L2-hot)
  gemm_bf16_wmma<true, false, true><<<grid, block, 0, stream>>>(
      adj, (const void*)supportT, bias, d_out, Nrows, OUT_F, Nrows);
}
